// RNN_30966714204923
// MI455X (gfx1250) — matrix operand order repaired and checked by reading, not compiled, not run on hardware
//
#include <hip/hip_runtime.h>
#include <hip/hip_bf16.h>
#include <math.h>

typedef __attribute__((ext_vector_type(16))) __bf16 v16bf;
typedef __attribute__((ext_vector_type(4)))  __bf16 v4bf;
typedef __attribute__((ext_vector_type(2)))  __bf16 v2bf;
typedef __attribute__((ext_vector_type(8)))  float  v8f;

static constexpr int B = 64, T = 4096, D = 128, H = 256;
static constexpr int HP = H + 8;   // padded LDS rows -> 16 A rows on distinct banks
static constexpr int DP = D + 8;

static __device__ __forceinline__ float fast_tanh(float v) {
#if __has_builtin(__builtin_amdgcn_tanhf)
    return __builtin_amdgcn_tanhf(v);      // v_tanh_f32 (gfx1250 trans op)
#else
    return tanhf(v);
#endif
}

// Fused RNN scan:  h_t = tanh(x_t @ Wx + bx + h_{t-1} @ Wh + bh)
// 4 workgroups (one per 16-batch tile) x 512 threads (16 wave32s).
// Wave w owns output columns [16w,16w+16): Wh (8 frags) + Wx (4 frags) in VGPRs.
// h and bf16-converted x both ping-pong in LDS; one barrier per timestep.
__global__ __launch_bounds__(512)
void rnn_scan_kernel(const float* __restrict__ x,
                     const float* __restrict__ Wx,
                     const float* __restrict__ bx,
                     const float* __restrict__ Wh,
                     const float* __restrict__ bh,
                     float* __restrict__ out)
{
    __shared__ __align__(16) __bf16 hbuf[2][16][HP];   // ~17 KB
    __shared__ __align__(16) __bf16 xbuf[2][16][DP];   // ~8.7 KB

    const int tid  = threadIdx.x;
    const int lane = tid & 31;
    const int wave = tid >> 5;                // 0..15
    const int b0   = blockIdx.x * 16;         // batch tile base

    const int n   = wave * 16 + (lane & 15);  // output column (N)
    const int m   = lane & 15;                // A-matrix row (M)
    const int kbA = (lane < 16) ? 0 : 8;      // A K-base per half-wave
    const int kbB = (lane < 16) ? 0 : 16;     // unused: B fragments are built with kbA

    // ---- register-resident B fragments --------------------------------
    v16bf whB[8];
    #pragma unroll
    for (int kt = 0; kt < 8; ++kt) {
        #pragma unroll
        for (int j = 0; j < 16; ++j)
            whB[kt][j] = (__bf16)Wh[(kt * 32 + kbA + (j < 8 ? j : j + 8)) * H + n];
    }
    v16bf wxB[4];
    #pragma unroll
    for (int kt = 0; kt < 4; ++kt) {
        #pragma unroll
        for (int j = 0; j < 16; ++j)
            wxB[kt][j] = (__bf16)Wx[(kt * 32 + kbA + (j < 8 ? j : j + 8)) * H + n];
    }
    const float biasv = bx[n] + bh[n];

    // A-fragment builder (16-bit A layout: slots 0-7 -> K=kb+0..7, slots 8-15 -> K=16+kb+0..7)
    auto buildA = [&](const v2bf* row, int kt) {
        v16bf a;
        #pragma unroll
        for (int p = 0; p < 8; ++p) {
            const int koff = ((p < 4) ? 0 : 16) + kbA + 2 * (p & 3);
            const v2bf u = row[(kt * 32 + koff) >> 1];
            a[2 * p]     = u.x;
            a[2 * p + 1] = u.y;
        }
        return a;
    };

    // bf16 pack + 8B LDS store for x staging
    auto stageStore = [&](__bf16* dst, float4 v) {
        v4bf pk = { (__bf16)v.x, (__bf16)v.y, (__bf16)v.z, (__bf16)v.w };
        *(v4bf*)dst = pk;
    };

    // h0 = 0
    for (int i = tid; i < 16 * HP; i += 512)
        (&hbuf[0][0][0])[i] = (__bf16)0.0f;

    // stage x_0: wave w stages row w, lane l covers columns 4l..4l+3
    const float* xsrc = x + (size_t)(b0 + wave) * T * D + lane * 4;
    stageStore(&xbuf[0][wave][lane * 4], *(const float4*)xsrc);
    __syncthreads();

    const int mo = (lane >> 4) * 8;           // C/D row base for this half-wave

    int cur = 0;
    for (int t = 0; t < T; ++t) {
        const int nxt = cur ^ 1;

        // 1) issue next timestep's x load early (latency hides under WMMAs)
        float4 xv;
        const bool do_stage = (t + 1 < T);
        if (do_stage) xv = *(const float4*)(xsrc + (size_t)(t + 1) * D);
        if (t + 2 < T) __builtin_prefetch(xsrc + (size_t)(t + 2) * D, 0, 1);

        // 2) two independent accumulation chains (halve WMMA dep depth)
        v8f accA, accB;
        #pragma unroll
        for (int r = 0; r < 8; ++r) { accA[r] = biasv; accB[r] = 0.0f; }

        const v2bf* xrow = (const v2bf*)&xbuf[cur][m][0];
        #pragma unroll
        for (int kt = 0; kt < 2; ++kt)
            accA = __builtin_amdgcn_wmma_f32_16x16x32_bf16(
                false, buildA(xrow, kt), false, wxB[kt], (short)0, accA, false, false);
        #pragma unroll
        for (int kt = 2; kt < 4; ++kt)
            accB = __builtin_amdgcn_wmma_f32_16x16x32_bf16(
                false, buildA(xrow, kt), false, wxB[kt], (short)0, accB, false, false);

        const v2bf* hrow = (const v2bf*)&hbuf[cur][m][0];
        #pragma unroll
        for (int kt = 0; kt < 4; ++kt)
            accA = __builtin_amdgcn_wmma_f32_16x16x32_bf16(
                false, buildA(hrow, kt), false, whB[kt], (short)0, accA, false, false);
        #pragma unroll
        for (int kt = 4; kt < 8; ++kt)
            accB = __builtin_amdgcn_wmma_f32_16x16x32_bf16(
                false, buildA(hrow, kt), false, whB[kt], (short)0, accB, false, false);

        // 3) tanh, write h_t to output + LDS ping-pong
        #pragma unroll
        for (int r = 0; r < 8; ++r) {
            const float hv   = fast_tanh(accA[r] + accB[r]);
            const int   mout = mo + r;
            out[((size_t)(b0 + mout) * T + t) * H + n] = hv;
            hbuf[nxt][mout][n] = (__bf16)hv;
        }

        // 4) finish staging x_{t+1}
        if (do_stage) stageStore(&xbuf[nxt][wave][lane * 4], xv);

        __syncthreads();
        cur = nxt;
    }
}

extern "C" void kernel_launch(void* const* d_in, const int* in_sizes, int n_in,
                              void* d_out, int out_size, void* d_ws, size_t ws_size,
                              hipStream_t stream) {
    (void)in_sizes; (void)n_in; (void)out_size; (void)d_ws; (void)ws_size;
    const float* x  = (const float*)d_in[0];
    const float* Wx = (const float*)d_in[1];
    const float* bx = (const float*)d_in[2];
    const float* Wh = (const float*)d_in[3];
    const float* bh = (const float*)d_in[4];
    float* out = (float*)d_out;

    dim3 grid(B / 16);   // 4 batch tiles
    dim3 block(512);     // 16 wave32s
    rnn_scan_kernel<<<grid, block, 0, stream>>>(x, Wx, bx, Wh, bh, out);
}
